// SpatialNeighborEmbedding_31147102830891
// MI455X (gfx1250) — matrix operand order repaired and checked by reading, not compiled, not run on hardware
//
#include <hip/hip_runtime.h>
#include <hip/hip_bf16.h>

// ---------------------------------------------------------------------------
// PointNet set abstraction for MI455X (gfx1250, wave32, WMMA bf16).
//   B=8, N=4096, C=3, F=16, NPOINT=1024, K=32, MLP 19->64->128->128, max over feat
// ---------------------------------------------------------------------------

constexpr int B_  = 8;
constexpr int N_  = 4096;
constexpr int F_  = 16;
constexpr int NPOINT_ = 1024;
constexpr int K_  = 32;

typedef __attribute__((ext_vector_type(16))) __bf16 v16bf;
typedef __attribute__((ext_vector_type(8)))  float  v8f;

static __device__ __forceinline__ __bf16 f2bf(float f) {
    return static_cast<__bf16>(f);            // native gfx1250 f32->bf16 (RNE)
}
static __device__ __forceinline__ unsigned short bf_bits(__bf16 b) {
    union { __bf16 b; unsigned short s; } x; x.b = b; return x.s;
}
static __device__ __forceinline__ __bf16 bits2bf(unsigned short s) {
    union { unsigned short s; __bf16 b; } x; x.s = s; return x.b;
}

// ---------------------------------------------------------------------------
// Kernel 1: farthest point sampling, one block per batch. Writes new_xyz.
// ---------------------------------------------------------------------------
__global__ __launch_bounds__(256)
void fps_kernel(const float* __restrict__ xyz, float* __restrict__ new_xyz) {
    const int b   = blockIdx.x;
    const int tid = threadIdx.x;
    const float* pts = xyz + (size_t)b * N_ * 3;

    __shared__ float dist[N_];
    __shared__ float rval[256];
    __shared__ int   ridx[256];

    for (int i = tid; i < N_; i += 256) dist[i] = 1e10f;
    __syncthreads();

    int far = 0;
    for (int s = 0; s < NPOINT_; ++s) {
        if (tid == 0) {
            float* o = new_xyz + ((size_t)b * NPOINT_ + s) * 3;
            o[0] = pts[far * 3 + 0];
            o[1] = pts[far * 3 + 1];
            o[2] = pts[far * 3 + 2];
        }
        const float fx = pts[far * 3 + 0];
        const float fy = pts[far * 3 + 1];
        const float fz = pts[far * 3 + 2];

        float best = -1.0f; int besti = 0;
        for (int i = tid; i < N_; i += 256) {
            float dx = pts[i * 3 + 0] - fx;
            float dy = pts[i * 3 + 1] - fy;
            float dz = pts[i * 3 + 2] - fz;
            float d  = dx * dx + dy * dy + dz * dz;
            float dm = fminf(dist[i], d);
            dist[i] = dm;
            if (dm > best) { best = dm; besti = i; }
        }
        rval[tid] = best; ridx[tid] = besti;
        __syncthreads();
        for (int off = 128; off > 0; off >>= 1) {
            if (tid < off) {
                float ov = rval[tid + off]; int oi = ridx[tid + off];
                if (ov > rval[tid] || (ov == rval[tid] && oi < ridx[tid])) {
                    rval[tid] = ov; ridx[tid] = oi;
                }
            }
            __syncthreads();
        }
        far = ridx[0];
        __syncthreads();
    }
}

// ---------------------------------------------------------------------------
// Kernel 2: KNN (K=32 incl. self). One thread per query; batch points tiled
// through LDS; 32-entry sorted insertion list (stable -> matches top_k ties).
// ---------------------------------------------------------------------------
__global__ __launch_bounds__(256)
void knn_kernel(const float* __restrict__ xyz, int* __restrict__ knn_idx) {
    const int b = blockIdx.y;
    const int q = blockIdx.x * 256 + threadIdx.x;
    const float* pts = xyz + (size_t)b * N_ * 3;

    const float qx = pts[q * 3 + 0];
    const float qy = pts[q * 3 + 1];
    const float qz = pts[q * 3 + 2];

    float kd[K_]; int ki[K_];
    #pragma unroll
    for (int i = 0; i < K_; ++i) { kd[i] = 3.0e38f; ki[i] = 0; }

    __shared__ float tile[1024 * 3];
    for (int t0 = 0; t0 < N_; t0 += 1024) {
        __syncthreads();
        for (int i = threadIdx.x; i < 1024 * 3; i += 256) tile[i] = pts[t0 * 3 + i];
        __syncthreads();
        for (int j = 0; j < 1024; ++j) {
            float dx = tile[j * 3 + 0] - qx;
            float dy = tile[j * 3 + 1] - qy;
            float dz = tile[j * 3 + 2] - qz;
            float d  = dx * dx + dy * dy + dz * dz;
            if (d < kd[K_ - 1]) {
                int p = K_ - 1;
                while (p > 0 && kd[p - 1] > d) {
                    kd[p] = kd[p - 1]; ki[p] = ki[p - 1]; --p;
                }
                kd[p] = d; ki[p] = t0 + j;
            }
        }
    }
    int* o = knn_idx + ((size_t)b * N_ + q) * K_;
    #pragma unroll
    for (int i = 0; i < K_; ++i) o[i] = ki[i];
}

// ---------------------------------------------------------------------------
// Kernel 3: weight prep. Transpose to Wt[out][K] bf16, K padded to mult of 32
// so each lane's WMMA B-fragment is 16 contiguous bf16 (32 bytes).
// ---------------------------------------------------------------------------
__global__ __launch_bounds__(256)
void prep_weights(const float* __restrict__ w1, const float* __restrict__ w2,
                  const float* __restrict__ w3,
                  unsigned short* __restrict__ Wt1,
                  unsigned short* __restrict__ Wt2,
                  unsigned short* __restrict__ Wt3) {
    const int t = blockIdx.x * 256 + threadIdx.x;
    if (t < 64 * 32) {                 // Wt1[64][32], w1 is [19][64]
        int n = t / 32, k = t % 32;
        Wt1[t] = (k < 19) ? bf_bits(f2bf(w1[k * 64 + n])) : (unsigned short)0;
    }
    if (t < 128 * 64) {                // Wt2[128][64], w2 is [64][128]
        int n = t / 64, k = t % 64;
        Wt2[t] = bf_bits(f2bf(w2[k * 128 + n]));
    }
    if (t < 128 * 128) {               // Wt3[128][128], w3 is [128][128]
        int n = t / 128, k = t % 128;
        Wt3[t] = bf_bits(f2bf(w3[k * 128 + n]));
    }
}

// ---------------------------------------------------------------------------
// Kernel 4: fused gather + MLP (WMMA bf16, f32 accum) + feature-max.
// ONE 16-row tile per wave (rows share (b,n); k in [k0,k0+16)) so every
// weight B-fragment is load-then-consume: no LICM hoisting, no spills.
// 8 waves/block, grid 8192 -> 65536 tiles.
// ---------------------------------------------------------------------------
__global__ __launch_bounds__(256)
void mlp_kernel(const float* __restrict__ xyz, const float* __restrict__ feat,
                const int* __restrict__ knn,
                const unsigned short* __restrict__ Wt1,
                const unsigned short* __restrict__ Wt2,
                const unsigned short* __restrict__ Wt3,
                const float* __restrict__ b1, const float* __restrict__ b2,
                const float* __restrict__ b3, float* __restrict__ outf) {
    __shared__ unsigned short h1[8][16][64];    // per-wave activation staging
    __shared__ unsigned short h2[8][16][128];

    const int lane = threadIdx.x & 31;
    const int wave = threadIdx.x >> 5;
    const int row  = lane & 15;        // A/B fragment row/col index
    const int half = lane >> 4;        // lane-half (ISA 16-bit A/B layout)
    const int klo  = half ? 16 : 0;    // A-frag K base within a 32-chunk
    const int kbB  = half ? 16 : 0;    // B-frag K base within a 32-chunk

    const int t    = blockIdx.x * 8 + wave;      // tile id
    const int rho0 = t * 16;                     // flat row base in [B,N,K]
    const int b    = rho0 / (N_ * K_);
    const int rem  = rho0 & (N_ * K_ - 1);
    const int n    = rem >> 5;
    const int k0   = rem & (K_ - 1);             // 0 or 16

    // ---- gather A1 (16x32 bf16, cols 19..31 zero) straight into A layout
    const int   j  = knn[((size_t)b * N_ + n) * K_ + k0 + row];
    const float* px = xyz  + ((size_t)b * N_ + j) * 3;
    const float* pf = feat + ((size_t)b * N_ + j) * F_;
    v16bf a1;
    #pragma unroll
    for (int i = 0; i < 8; ++i) {
        int q0 = klo + i, q1 = klo + 8 + i;
        float v0 = (q0 < 3) ? px[q0] : ((q0 < 19) ? pf[q0 - 3] : 0.0f);
        float v1 = (q1 < 3) ? px[q1] : ((q1 < 19) ? pf[q1 - 3] : 0.0f);
        a1[i]     = f2bf(v0);
        a1[8 + i] = f2bf(v1);
    }

    // ---- layer 1: 19(pad 32) -> 64, relu, stage to LDS
    #pragma unroll
    for (int nt = 0; nt < 4; ++nt) {
        const int ncol = nt * 16 + row;
        const unsigned short* wr = Wt1 + ncol * 32 + kbB;
        v16bf bf;
        #pragma unroll
        for (int i = 0; i < 16; ++i) bf[i] = bits2bf(wr[i]);
        const float bias = b1[ncol];
        v8f c;
        #pragma unroll
        for (int r = 0; r < 8; ++r) c[r] = bias;
        c = __builtin_amdgcn_wmma_f32_16x16x32_bf16(false, a1, false, bf,
                                                    (short)0, c, false, false);
        #pragma unroll
        for (int r = 0; r < 8; ++r)
            h1[wave][half * 8 + r][ncol] = bf_bits(f2bf(fmaxf(c[r], 0.0f)));
    }

    // ---- A2 fragments (16x64 -> 2 K-chunks)
    v16bf a2[2];
    #pragma unroll
    for (int cc = 0; cc < 2; ++cc) {
        const int kb = cc * 32 + klo;
        #pragma unroll
        for (int i = 0; i < 8; ++i) {
            a2[cc][i]     = bits2bf(h1[wave][row][kb + i]);
            a2[cc][8 + i] = bits2bf(h1[wave][row][kb + 8 + i]);
        }
    }

    // ---- layer 2: 64 -> 128, relu, stage to LDS
    #pragma unroll
    for (int nt = 0; nt < 8; ++nt) {
        const int ncol = nt * 16 + row;
        const float bias = b2[ncol];
        v8f c;
        #pragma unroll
        for (int r = 0; r < 8; ++r) c[r] = bias;
        #pragma unroll
        for (int cc = 0; cc < 2; ++cc) {
            const unsigned short* wr = Wt2 + ncol * 64 + cc * 32 + kbB;
            v16bf bf;
            #pragma unroll
            for (int i = 0; i < 16; ++i) bf[i] = bits2bf(wr[i]);
            c = __builtin_amdgcn_wmma_f32_16x16x32_bf16(false, a2[cc], false, bf,
                                                        (short)0, c, false, false);
        }
        #pragma unroll
        for (int r = 0; r < 8; ++r)
            h2[wave][half * 8 + r][ncol] = bf_bits(f2bf(fmaxf(c[r], 0.0f)));
    }

    // ---- A3 fragments (16x128 -> 4 K-chunks)
    v16bf a3[4];
    #pragma unroll
    for (int cc = 0; cc < 4; ++cc) {
        const int kb = cc * 32 + klo;
        #pragma unroll
        for (int i = 0; i < 8; ++i) {
            a3[cc][i]     = bits2bf(h2[wave][row][kb + i]);
            a3[cc][8 + i] = bits2bf(h2[wave][row][kb + 8 + i]);
        }
    }

    // ---- layer 3: 128 -> 128 (no relu) + running row-max over columns
    float rmax[8];
    #pragma unroll
    for (int r = 0; r < 8; ++r) rmax[r] = -3.0e38f;
    #pragma unroll
    for (int nt = 0; nt < 8; ++nt) {
        const int ncol = nt * 16 + row;
        const float bias = b3[ncol];
        v8f c;
        #pragma unroll
        for (int r = 0; r < 8; ++r) c[r] = bias;
        #pragma unroll
        for (int cc = 0; cc < 4; ++cc) {
            const unsigned short* wr = Wt3 + ncol * 128 + cc * 32 + kbB;
            v16bf bf;
            #pragma unroll
            for (int i = 0; i < 16; ++i) bf[i] = bits2bf(wr[i]);
            c = __builtin_amdgcn_wmma_f32_16x16x32_bf16(false, a3[cc], false, bf,
                                                        (short)0, c, false, false);
        }
        #pragma unroll
        for (int r = 0; r < 8; ++r) rmax[r] = fmaxf(rmax[r], c[r]);
    }

    // ---- reduce max across the 16 lanes of each half (lane = column)
    #pragma unroll
    for (int off = 1; off < 16; off <<= 1) {
        #pragma unroll
        for (int r = 0; r < 8; ++r)
            rmax[r] = fmaxf(rmax[r], __shfl_xor(rmax[r], off, 32));
    }
    float* op = outf + rho0;
    if (lane == 0) {
        #pragma unroll
        for (int r = 0; r < 8; ++r) op[r] = rmax[r];       // rows 0..7
    }
    if (lane == 16) {
        #pragma unroll
        for (int r = 0; r < 8; ++r) op[8 + r] = rmax[r];   // rows 8..15
    }
}

// ---------------------------------------------------------------------------
extern "C" void kernel_launch(void* const* d_in, const int* in_sizes, int n_in,
                              void* d_out, int out_size, void* d_ws, size_t ws_size,
                              hipStream_t stream) {
    const float* xyz  = (const float*)d_in[0];
    const float* feat = (const float*)d_in[1];
    const float* w1   = (const float*)d_in[2];
    const float* b1   = (const float*)d_in[3];
    const float* w2   = (const float*)d_in[4];
    const float* b2   = (const float*)d_in[5];
    const float* w3   = (const float*)d_in[6];
    const float* b3   = (const float*)d_in[7];

    float* new_xyz  = (float*)d_out;                             // [8,1024,3]
    float* new_feat = (float*)d_out + (size_t)B_ * NPOINT_ * 3;  // [8,4096,32]

    // workspace layout: knn_idx (4 MB) | Wt1 bf16 | Wt2 bf16 | Wt3 bf16
    char* ws = (char*)d_ws;
    int*            knn_idx = (int*)ws;
    unsigned short* Wt1 = (unsigned short*)(ws + (size_t)B_ * N_ * K_ * 4);
    unsigned short* Wt2 = Wt1 + 64 * 32;
    unsigned short* Wt3 = Wt2 + 128 * 64;

    fps_kernel<<<B_, 256, 0, stream>>>(xyz, new_xyz);
    knn_kernel<<<dim3(N_ / 256, B_), 256, 0, stream>>>(xyz, knn_idx);
    prep_weights<<<(128 * 128 + 255) / 256, 256, 0, stream>>>(w1, w2, w3, Wt1, Wt2, Wt3);

    // 65536 16-row tiles / (8 waves per block) = 8192 blocks
    mlp_kernel<<<8192, 256, 0, stream>>>(xyz, feat, knn_idx, Wt1, Wt2, Wt3,
                                         b1, b2, b3, new_feat);
}
